// ContrastiveWithEntropyLoss_5738076307780
// MI455X (gfx1250) — matrix operand order repaired and checked by reading, not compiled, not run on hardware
//
#include <hip/hip_runtime.h>
#include <hip/hip_bf16.h>
#include <math.h>

#define BATCH   16384
#define CLS     2048
#define NROW    4096      // 2*CLS
#define EPS_F   1e-8f

typedef __bf16 v8bf  __attribute__((ext_vector_type(8)));
typedef __bf16 v16bf __attribute__((ext_vector_type(16)));
typedef float  v8f   __attribute__((ext_vector_type(8)));

// Match the builtin's parameter types exactly (per hipcc diagnostic):
// 'int __attribute__((vector_size(16))) __device__ *' for the global source.
typedef int mv4i __attribute__((vector_size(4 * sizeof(int))));
typedef __attribute__((address_space(1))) mv4i as1_v4i;
typedef __attribute__((address_space(3))) mv4i as3_v4i;

#if __has_builtin(__builtin_amdgcn_global_load_async_to_lds_b128) && \
    __has_builtin(__builtin_amdgcn_s_wait_asynccnt)
#define GEMM_ASYNC 1
#else
#define GEMM_ASYNC 0
#endif

static __device__ __forceinline__ __bf16 f2bf(float x){
  unsigned u = __builtin_bit_cast(unsigned, x);
  unsigned r = u + 0x7FFFu + ((u >> 16) & 1u);
  unsigned short s = (unsigned short)(r >> 16);
  return __builtin_bit_cast(__bf16, s);
}
static __device__ __forceinline__ float bf2f(__bf16 h){
  unsigned short s = __builtin_bit_cast(unsigned short, h);
  unsigned u = ((unsigned)s) << 16;
  return __builtin_bit_cast(float, u);
}
static __device__ __forceinline__ v16bf cat8(v8bf a, v8bf b){
  return __builtin_shufflevector(a, b, 0,1,2,3,4,5,6,7,8,9,10,11,12,13,14,15);
}

// 16B tile-chunk copy global -> LDS.
// Async path: GLOBAL_LOAD_ASYNC_TO_LDS_B128, tracked by ASYNCcnt (no VGPR round trip).
// AS pointers built via integer casts: low 32 bits of a generic LDS pointer are the
// LDS byte offset (CDNA5 aperture rule); generic global pointer value == AS1 address.
static __device__ __forceinline__ void cp16(__bf16* dst_lds, const __bf16* src_g){
#if GEMM_ASYNC
  as3_v4i* d = (as3_v4i*)(unsigned long long)(unsigned)(unsigned long long)dst_lds;
  as1_v4i* s = (as1_v4i*)(unsigned long long)src_g;
  __builtin_amdgcn_global_load_async_to_lds_b128(s, d, 0, 0);
#else
  *(uint4*)dst_lds = *(const uint4*)src_g;
#endif
}
static __device__ __forceinline__ void cp_wait_keep8(){
#if GEMM_ASYNC
  __builtin_amdgcn_s_wait_asynccnt(8);
#endif
}
static __device__ __forceinline__ void cp_wait_all(){
#if GEMM_ASYNC
  __builtin_amdgcn_s_wait_asynccnt(0);
#endif
}

// ---------------------------------------------------------------------------
// 1) Transpose q_i/q_j columns into q[N, B] rows, split fp32 -> bf16 hi + lo.
// ---------------------------------------------------------------------------
__global__ __launch_bounds__(256)
void k_transpose(const float* __restrict__ qi, const float* __restrict__ qj,
                 __bf16* __restrict__ qh, __bf16* __restrict__ ql){
  __shared__ float t[32][33];
  const int tx = threadIdx.x, ty = threadIdx.y;
  const int b0 = blockIdx.x * 32;
  const int by = blockIdx.y;                 // 0..127 (cluster-row block)
  const float* src = (by < 64) ? qi : qj;
  const int c0s = (by & 63) * 32;            // column base in source matrix
  const int n0  = by * 32;                   // cluster row base in output
#pragma unroll
  for (int i = 0; i < 4; ++i){
    int r = ty + 8*i;
    t[r][tx] = src[(size_t)(b0 + r) * CLS + c0s + tx];
  }
  __syncthreads();
#pragma unroll
  for (int i = 0; i < 4; ++i){
    int c = ty + 8*i;
    float x = t[tx][c];
    __bf16 hi = f2bf(x);
    __bf16 lo = f2bf(x - bf2f(hi));
    size_t o = (size_t)(n0 + c) * BATCH + b0 + tx;
    qh[o] = hi; ql[o] = lo;
  }
}

// ---------------------------------------------------------------------------
// 2) Column sums + sum-of-squares, stage 1 (partials, deterministic).
// ---------------------------------------------------------------------------
__global__ __launch_bounds__(256)
void k_colpart(const float* __restrict__ qi, const float* __restrict__ qj,
               float* __restrict__ psum, float* __restrict__ pssq){
  const int col = blockIdx.x * 256 + threadIdx.x;   // 0..4095
  const int rb  = blockIdx.y;                       // 0..63
  const float* src = (col < CLS) ? qi : qj;
  const int cc = col & (CLS - 1);
  float s = 0.f, q = 0.f;
  const int b0 = rb * (BATCH/64);
  for (int i = 0; i < BATCH/64; ++i){
    float x = src[(size_t)(b0 + i) * CLS + cc];
    s += x; q += x*x;
  }
  psum[rb * NROW + col] = s;
  pssq[rb * NROW + col] = q;
}

// 2b) stage 2: fold 64 partials -> colsum, norms
__global__ __launch_bounds__(256)
void k_colfin(const float* __restrict__ psum, const float* __restrict__ pssq,
              float* __restrict__ colsum, float* __restrict__ norms){
  const int col = blockIdx.x * 256 + threadIdx.x;
  float s = 0.f, q = 0.f;
  for (int rb = 0; rb < 64; ++rb){
    s += psum[rb*NROW + col];
    q += pssq[rb*NROW + col];
  }
  colsum[col] = s;
  norms[col]  = sqrtf(q);
}

// ---------------------------------------------------------------------------
// 3) Entropy of cluster marginals (single WG, tree-reduced, deterministic).
// ---------------------------------------------------------------------------
__global__ __launch_bounds__(256)
void k_entropy(const float* __restrict__ colsum, float* __restrict__ ent){
  __shared__ float red[256];
  const int tx = threadIdx.x;
  float sci = 0.f, sli = 0.f, scj = 0.f, slj = 0.f;
  for (int c = tx; c < CLS; c += 256){
    float a = colsum[c];        sci += a; sli += a * logf(a);
    float b = colsum[c + CLS];  scj += b; slj += b * logf(b);
  }
  float vals[4] = {sci, sli, scj, slj};
  float tot[4];
  for (int k = 0; k < 4; ++k){
    red[tx] = vals[k]; __syncthreads();
    for (int off = 128; off > 0; off >>= 1){
      if (tx < off) red[tx] += red[tx + off];
      __syncthreads();
    }
    tot[k] = red[0]; __syncthreads();
  }
  if (tx == 0){
    // sum p log p = (sum c log c)/tot - log(tot)
    float ne_i = logf((float)CLS) + tot[1]/tot[0] - logf(tot[0]);
    float ne_j = logf((float)CLS) + tot[3]/tot[2] - logf(tot[2]);
    ent[0] = ne_i + ne_j;
  }
}

// ---------------------------------------------------------------------------
// 4) Gram matrix G = q q^T via bf16 WMMA with hi/lo split (3 products).
//    WG tile 128x128, 8 waves (4M x 2N), K stepped 32 through LDS,
//    double-buffered with async global->LDS staging.
// ---------------------------------------------------------------------------
#define LDS_STRIDE 40   // 32 K elems + 8 pad (80B rows: conflict-free frag reads)
#define KSTEPS (BATCH/32)

__global__ __launch_bounds__(256)
void k_gemm(const __bf16* __restrict__ qh, const __bf16* __restrict__ ql,
            float* __restrict__ G){
  __shared__ __align__(16) __bf16 sAh[2][128*LDS_STRIDE];
  __shared__ __align__(16) __bf16 sAl[2][128*LDS_STRIDE];
  __shared__ __align__(16) __bf16 sBh[2][128*LDS_STRIDE];
  __shared__ __align__(16) __bf16 sBl[2][128*LDS_STRIDE];

  const int tid  = threadIdx.x;
  const int lane = tid & 31;
  const int wave = tid >> 5;
  const int wm   = wave & 3;      // M offset = wm*32
  const int wn   = wave >> 2;     // N offset = wn*64
  const int l15  = lane & 15;
  const int lhi  = lane >> 4;
  const int rowbase = blockIdx.y * 128;
  const int colbase = blockIdx.x * 128;

  // Per-thread staging chunks: tile = 128 rows x 32 K bf16 = 512 x 16B chunks,
  // thread owns chunks tid and tid+256.
  const int rr0 = tid >> 2, qq8 = (tid & 3) * 8;
  const int rr1 = rr0 + 64;
  const int lo0 = rr0*LDS_STRIDE + qq8;
  const int lo1 = rr1*LDS_STRIDE + qq8;
  const __bf16* gAh0 = qh + (size_t)(rowbase + rr0) * BATCH + qq8;
  const __bf16* gAh1 = qh + (size_t)(rowbase + rr1) * BATCH + qq8;
  const __bf16* gAl0 = ql + (size_t)(rowbase + rr0) * BATCH + qq8;
  const __bf16* gAl1 = ql + (size_t)(rowbase + rr1) * BATCH + qq8;
  const __bf16* gBh0 = qh + (size_t)(colbase + rr0) * BATCH + qq8;
  const __bf16* gBh1 = qh + (size_t)(colbase + rr1) * BATCH + qq8;
  const __bf16* gBl0 = ql + (size_t)(colbase + rr0) * BATCH + qq8;
  const __bf16* gBl1 = ql + (size_t)(colbase + rr1) * BATCH + qq8;

  auto issue = [&](int buf, int kb){
    const size_t ko = (size_t)kb * 32;
    cp16(&sAh[buf][lo0], gAh0 + ko);
    cp16(&sAh[buf][lo1], gAh1 + ko);
    cp16(&sAl[buf][lo0], gAl0 + ko);
    cp16(&sAl[buf][lo1], gAl1 + ko);
    cp16(&sBh[buf][lo0], gBh0 + ko);
    cp16(&sBh[buf][lo1], gBh1 + ko);
    cp16(&sBl[buf][lo0], gBl0 + ko);
    cp16(&sBl[buf][lo1], gBl1 + ko);
  };

  v8f acc[2][4];
#pragma unroll
  for (int mt = 0; mt < 2; ++mt)
#pragma unroll
    for (int nt = 0; nt < 4; ++nt)
      acc[mt][nt] = (v8f)(0.0f);

  issue(0, 0);
  for (int kb = 0; kb < KSTEPS; ++kb){
    const int cur = kb & 1;
    if (kb + 1 < KSTEPS){
      issue(cur ^ 1, kb + 1);   // prefetch next K-step into the other buffer
      cp_wait_keep8();          // oldest 8 (current buffer) have landed
    } else {
      cp_wait_all();
    }
    __syncthreads();
#pragma unroll
    for (int mt = 0; mt < 2; ++mt){
      // A frag (16x32 bf16): lanes 0-15 carry K 0..7 & 16..23, lanes 16-31 K 8..15 & 24..31
      int ar = (wm*32 + mt*16 + l15)*LDS_STRIDE + lhi*8;
      v16bf a_h = cat8(*(const v8bf*)&sAh[cur][ar], *(const v8bf*)&sAh[cur][ar + 16]);
      v16bf a_l = cat8(*(const v8bf*)&sAl[cur][ar], *(const v8bf*)&sAl[cur][ar + 16]);
#pragma unroll
      for (int nt = 0; nt < 4; ++nt){
        // B frag (32x16 bf16): lane = N col, lanes 0-15 K 0..7 & 16..23, lanes 16-31 K 8..15 & 24..31
        int br = (wn*64 + nt*16 + l15)*LDS_STRIDE + lhi*8;
        v16bf b_h = cat8(*(const v8bf*)&sBh[cur][br], *(const v8bf*)&sBh[cur][br + 16]);
        v16bf b_l = cat8(*(const v8bf*)&sBl[cur][br], *(const v8bf*)&sBl[cur][br + 16]);
        acc[mt][nt] = __builtin_amdgcn_wmma_f32_16x16x32_bf16(
            false, a_h, false, b_h, (short)0, acc[mt][nt], false, false);
        acc[mt][nt] = __builtin_amdgcn_wmma_f32_16x16x32_bf16(
            false, a_h, false, b_l, (short)0, acc[mt][nt], false, false);
        acc[mt][nt] = __builtin_amdgcn_wmma_f32_16x16x32_bf16(
            false, a_l, false, b_h, (short)0, acc[mt][nt], false, false);
      }
    }
    __syncthreads();
  }
  // C/D layout: lanes 0-15: (M=v, N=lane); lanes 16-31: (M=8+v, N=lane-16)
#pragma unroll
  for (int mt = 0; mt < 2; ++mt)
#pragma unroll
    for (int nt = 0; nt < 4; ++nt)
#pragma unroll
      for (int v = 0; v < 8; ++v){
        int row = rowbase + wm*32 + mt*16 + lhi*8 + v;
        int col = colbase + wn*64 + nt*16 + l15;
        G[(size_t)row * NROW + col] = acc[mt][nt][v];
      }
}

// ---------------------------------------------------------------------------
// 5) Per-row: normalize, stable logsumexp excluding diagonal, minus positive.
// ---------------------------------------------------------------------------
__global__ __launch_bounds__(256)
void k_lse(const float* __restrict__ G, const float* __restrict__ norms,
           float* __restrict__ rowloss){
  __shared__ float red[256];
  __shared__ float shPos;
  const int r  = blockIdx.x;
  const int tx = threadIdx.x;
  const float nr = norms[r];
  float sv[16];
  float m = -INFINITY;
#pragma unroll
  for (int j = 0; j < 16; ++j){
    int k = j*256 + tx;
    float g = G[(size_t)r * NROW + k];
    float d = fmaxf(nr * norms[k], EPS_F);
    float s = g / d;                 // TEMPERATURE == 1
    if (k == r) s = -INFINITY;       // exclude diagonal
    sv[j] = s;
    m = fmaxf(m, s);
  }
  const int pk = r ^ CLS;            // positive partner column
  if ((pk & 255) == tx) shPos = sv[pk >> 8];
  red[tx] = m; __syncthreads();
  for (int off = 128; off > 0; off >>= 1){
    if (tx < off) red[tx] = fmaxf(red[tx], red[tx + off]);
    __syncthreads();
  }
  m = red[0]; __syncthreads();
  float a = 0.f;
#pragma unroll
  for (int j = 0; j < 16; ++j) a += expf(sv[j] - m);   // exp(-inf)=0 handles diag
  red[tx] = a; __syncthreads();
  for (int off = 128; off > 0; off >>= 1){
    if (tx < off) red[tx] += red[tx + off];
    __syncthreads();
  }
  if (tx == 0) rowloss[r] = m + logf(red[0]) - shPos;
}

// ---------------------------------------------------------------------------
// 6) Final scalar: mean row loss + entropy.
// ---------------------------------------------------------------------------
__global__ __launch_bounds__(256)
void k_final(const float* __restrict__ rowloss, const float* __restrict__ ent,
             float* __restrict__ out){
  __shared__ float red[256];
  const int tx = threadIdx.x;
  float a = 0.f;
  for (int j = 0; j < NROW/256; ++j) a += rowloss[j*256 + tx];
  red[tx] = a; __syncthreads();
  for (int off = 128; off > 0; off >>= 1){
    if (tx < off) red[tx] += red[tx + off];
    __syncthreads();
  }
  if (tx == 0) out[0] = red[0] / (float)NROW + ent[0];
}

// ---------------------------------------------------------------------------
extern "C" void kernel_launch(void* const* d_in, const int* in_sizes, int n_in,
                              void* d_out, int out_size, void* d_ws, size_t ws_size,
                              hipStream_t stream){
  const float* qi = (const float*)d_in[0];
  const float* qj = (const float*)d_in[1];
  float* out = (float*)d_out;
  char* ws = (char*)d_ws;

  const size_t offQH = 0;                                    // 128 MB
  const size_t offQL = offQH + (size_t)NROW * BATCH * 2;     // 128 MB
  const size_t offG  = offQL + (size_t)NROW * BATCH * 2;     // 64 MB
  const size_t offPS = offG  + (size_t)NROW * NROW * 4;      // 1 MB
  const size_t offPQ = offPS + (size_t)64 * NROW * 4;        // 1 MB
  const size_t offCS = offPQ + (size_t)64 * NROW * 4;        // 16 KB
  const size_t offNM = offCS + (size_t)NROW * 4;             // 16 KB
  const size_t offRL = offNM + (size_t)NROW * 4;             // 16 KB
  const size_t offEN = offRL + (size_t)NROW * 4;             // 4 B

  __bf16* qh     = (__bf16*)(ws + offQH);
  __bf16* ql     = (__bf16*)(ws + offQL);
  float*  G      = (float*)(ws + offG);
  float*  psum   = (float*)(ws + offPS);
  float*  pssq   = (float*)(ws + offPQ);
  float*  colsum = (float*)(ws + offCS);
  float*  norms  = (float*)(ws + offNM);
  float*  rloss  = (float*)(ws + offRL);
  float*  ent    = (float*)(ws + offEN);

  k_transpose<<<dim3(BATCH/32, NROW/32), dim3(32, 8), 0, stream>>>(qi, qj, qh, ql);
  k_colpart  <<<dim3(NROW/256, 64), 256, 0, stream>>>(qi, qj, psum, pssq);
  k_colfin   <<<NROW/256, 256, 0, stream>>>(psum, pssq, colsum, norms);
  k_entropy  <<<1, 256, 0, stream>>>(colsum, ent);
  k_gemm     <<<dim3(NROW/128, NROW/128), 256, 0, stream>>>(qh, ql, G);
  k_lse      <<<NROW, 256, 0, stream>>>(G, norms, rloss);
  k_final    <<<1, 256, 0, stream>>>(rloss, ent, out);
}
